// AttentionUnit_89532888253172
// MI455X (gfx1250) — matrix operand order repaired and checked by reading, not compiled, not run on hardware
//
#include <hip/hip_runtime.h>
#include <hip/hip_bf16.h>
#include <cstdint>
#include <cstddef>

// DIN attention unit, CDNA5 (gfx1250), wave32, bf16 WMMA 16x16x32.
// B=4096, T=200 (pad 208), E=64, H1=80, H2=40.
// LDS pitches padded (136 / 104 bf16) for conflict-free ds_load_b128 fragments.

typedef __attribute__((ext_vector_type(16))) __bf16 v16bf;
typedef __attribute__((ext_vector_type(8)))  float  v8f;

#define T_      200
#define E_      64
#define H1_     80
#define H2_     40
#define TP_     208
#define NTILES_ 13
#define CH_RT   3        // row tiles (of 16) per chunk
#define CH_ROWS 48
#define PA_     136      // sA / sWeffT pitch (bf16)
#define PH_     104      // sH1 / sW2T pitch (bf16)

// ws layout: [0,20480) WeffT bf16[80][128] | [20480,40960) Wq f32[80][64] | [40960,50176) W2T bf16[48][96]

union Frag16 { v16bf v; uint4 q[2]; };

static __device__ __forceinline__ v16bf load_frag(const __bf16* p0, const __bf16* p1) {
  Frag16 f;
  f.q[0] = *reinterpret_cast<const uint4*>(p0);
  f.q[1] = *reinterpret_cast<const uint4*>(p1);
  return f.v;
}

static __device__ __forceinline__ unsigned pack_bf2(float x, float y) {
  union { __bf16 h[2]; unsigned u; } p;
  p.h[0] = (__bf16)x; p.h[1] = (__bf16)y;
  return p.u;
}

__global__ __launch_bounds__(256) void din_prep(
    const float* __restrict__ W1, const float* __restrict__ W2,
    __bf16* __restrict__ wEffT, float* __restrict__ wQ, __bf16* __restrict__ w2T)
{
  const int tid = threadIdx.x;
  // WeffT[h][k]: k<64 -> W1[:,64:128]-W1[:,128:192] ; k>=64 -> W1[:,192:256]
  for (int i = tid; i < 80 * 128; i += 256) {
    int h = i / 128, k = i % 128;
    float v = (k < 64) ? (W1[h * 256 + 64 + k] - W1[h * 256 + 128 + k])
                       : W1[h * 256 + 192 + (k - 64)];
    wEffT[i] = (__bf16)v;
  }
  // Wq[h][e] = W1[:,0:64] + W1[:,128:192]  (folded into per-batch bias)
  for (int i = tid; i < 80 * 64; i += 256) {
    int h = i / 64, e = i % 64;
    wQ[i] = W1[h * 256 + e] + W1[h * 256 + 128 + e];
  }
  // W2T[n][k] = W2[n][k], zero-padded to 48x96
  for (int i = tid; i < 48 * 96; i += 256) {
    int n = i / 96, k = i % 96;
    w2T[i] = (k < 80 && n < 40) ? (__bf16)W2[n * 80 + k] : (__bf16)0.0f;
  }
}

__global__ __launch_bounds__(256) void din_attn(
    const float* __restrict__ query, const float* __restrict__ keys,
    const int* __restrict__ klen,
    const float* __restrict__ b1, const float* __restrict__ a1p,
    const float* __restrict__ b2, const float* __restrict__ a2p,
    const float* __restrict__ W3, const float* __restrict__ b3p,
    const __bf16* __restrict__ wEffT, const float* __restrict__ wQ,
    const __bf16* __restrict__ w2T,
    float* __restrict__ out)
{
  __shared__ __align__(16) __bf16 sWeffT[80][PA_];   // B GEMM1, [N][K] transposed
  __shared__ __align__(16) __bf16 sW2T[48][PH_];     // B GEMM2, [N][K] transposed
  __shared__ __align__(16) __bf16 sA[CH_ROWS][PA_];  // A chunk [t][k|q*k]
  __shared__ __align__(16) __bf16 sH1[CH_ROWS][PH_]; // h1 chunk (A for GEMM2)
  __shared__ float sH2[CH_ROWS][H2_];
  __shared__ float sQ[64];
  __shared__ float sC1[80];     // per-batch folded bias
  __shared__ float sScore[TP_];
  __shared__ float sRed[256];

  const int tid  = threadIdx.x;
  const int b    = blockIdx.x;
  const int lane = tid & 31;
  const int wave = tid >> 5;
  const int m    = lane & 15;   // row/col within 16 for frags
  const int hi   = lane >> 4;   // half-wave select

  // stage weights into LDS (row-wise 128-bit copies into padded pitch)
  {
    const uint4* s0 = reinterpret_cast<const uint4*>(wEffT);     // [80][128] -> 8 uint4/row
    for (int i = tid; i < 80 * 8; i += 256) {
      int n = i >> 3, c = i & 7;
      *reinterpret_cast<uint4*>(&sWeffT[n][c * 8]) = s0[i];
    }
    const uint4* s1 = reinterpret_cast<const uint4*>(w2T);       // [48][96] -> 6 uint4/row
    for (int i = tid; i < 48 * 6; i += 256) {
      int n = i / 6, c = i % 6;
      *reinterpret_cast<uint4*>(&sW2T[n][c * 8]) = s1[i];
    }
  }
  if (tid < 64) sQ[tid] = query[(size_t)b * 64 + tid];
  // zero pad columns 80..95 of sH1 (K padding for GEMM2)
  for (int i = tid; i < CH_ROWS * 8; i += 256)
    *reinterpret_cast<unsigned*>(&sH1[i >> 3][80 + ((i & 7) << 1)]) = 0u;
  __syncthreads();

  // c1[h] = b1[h] + Wq[h,:] . q   (per-batch GEMV, 80x64)
  if (tid < 80) {
    float acc = b1[tid];
    const float* wr = wQ + tid * 64;
    #pragma unroll 8
    for (int e = 0; e < 64; ++e) acc += wr[e] * sQ[e];
    sC1[tid] = acc;
  }
  __syncthreads();

  const float a1v = a1p[0];
  const float a2v = a2p[0];
  const float b3v = b3p[0];
  const float* kb = keys + (size_t)b * T_ * E_;

  for (int ct = 0; ct < NTILES_; ct += CH_RT) {
    const int nrt  = (NTILES_ - ct < CH_RT) ? (NTILES_ - ct) : CH_RT;
    const int t0   = ct * 16;
    const int rows = nrt * 16;

    // build bf16 A chunk: [k | q*k], zero rows for t >= T; paired b32 stores
    for (int i = tid; i < rows * 32; i += 256) {
      int r = i >> 5, e2 = (i & 31) << 1;
      int t = t0 + r;
      float k0 = 0.0f, k1 = 0.0f;
      if (t < T_) {
        k0 = kb[(size_t)t * 64 + e2];
        k1 = kb[(size_t)t * 64 + e2 + 1];
      }
      *reinterpret_cast<unsigned*>(&sA[r][e2])      = pack_bf2(k0, k1);
      *reinterpret_cast<unsigned*>(&sA[r][64 + e2]) = pack_bf2(sQ[e2] * k0, sQ[e2 + 1] * k1);
    }
    // prefetch next chunk of keys (global_prefetch_b8)
    {
      int tpre = t0 + rows + (tid >> 2);
      if (tpre < T_ && (tid >> 2) < CH_ROWS)
        __builtin_prefetch(kb + (size_t)tpre * 64 + ((tid & 3) << 4), 0, 0);
    }
    __syncthreads();

    // ---- GEMM1: [rows x 128]bf16 x [128 x 80]bf16 -> h1, PReLU, store bf16
    {
      const int njobs = nrt * 5;
      for (int job = wave; job < njobs; job += 8) {
        const int rt  = job / 5, nt = job % 5;
        const int row = rt * 16 + m;
        const int n   = nt * 16 + m;
        v8f c = {};
        #pragma unroll
        for (int kk = 0; kk < 128; kk += 32) {
          v16bf a = load_frag(&sA[row][kk + 8 * hi], &sA[row][kk + 16 + 8 * hi]);
          v16bf w = load_frag(&sWeffT[n][kk + 8 * hi], &sWeffT[n][kk + 16 + 8 * hi]);
          c = __builtin_amdgcn_wmma_f32_16x16x32_bf16(false, a, false, w,
                                                      (short)0, c, false, false);
        }
        const float c1n = sC1[n];
        #pragma unroll
        for (int j = 0; j < 8; ++j) {        // C/D f32 16x16 layout
          int r2 = rt * 16 + j + 8 * hi;
          float v = c[j] + c1n;
          v = (v >= 0.0f) ? v : a1v * v;
          sH1[r2][n] = (__bf16)v;
        }
      }
    }
    __syncthreads();

    // ---- GEMM2: [rows x 96]bf16 x [96 x 48]bf16 -> h2, PReLU, store f32
    {
      const int njobs = nrt * 3;
      for (int job = wave; job < njobs; job += 8) {
        const int rt  = job / 3, nt = job % 3;
        const int row = rt * 16 + m;
        const int n   = nt * 16 + m;
        v8f c = {};
        #pragma unroll
        for (int kk = 0; kk < 96; kk += 32) {
          v16bf a = load_frag(&sH1[row][kk + 8 * hi], &sH1[row][kk + 16 + 8 * hi]);
          v16bf w = load_frag(&sW2T[n][kk + 8 * hi], &sW2T[n][kk + 16 + 8 * hi]);
          c = __builtin_amdgcn_wmma_f32_16x16x32_bf16(false, a, false, w,
                                                      (short)0, c, false, false);
        }
        if (n < H2_) {
          const float bias = b2[n];
          #pragma unroll
          for (int j = 0; j < 8; ++j) {
            int r2 = rt * 16 + j + 8 * hi;
            float v = c[j] + bias;
            v = (v >= 0.0f) ? v : a2v * v;
            sH2[r2][n] = v;
          }
        }
      }
    }
    __syncthreads();

    // scores: h2 . w3 + b3
    if (tid < rows) {
      float acc = b3v;
      #pragma unroll 8
      for (int n = 0; n < H2_; ++n) acc += sH2[tid][n] * W3[n];
      sScore[t0 + tid] = acc;
    }
    __syncthreads();
  }

  // ---- masked softmax over T (len==0 -> uniform, matching reference)
  const int len = klen[b];
  float s = -INFINITY;
  if (tid < TP_) {
    s = sScore[tid];
    if (tid >= T_)       s = -INFINITY;
    else if (tid >= len) s = -1e9f;
  }
  sRed[tid] = s;
  __syncthreads();
  for (int off = 128; off > 0; off >>= 1) {
    if (tid < off) sRed[tid] = fmaxf(sRed[tid], sRed[tid + off]);
    __syncthreads();
  }
  const float mx = sRed[0];
  __syncthreads();
  float ev = (tid < TP_) ? __expf(s - mx) : 0.0f;
  sRed[tid] = ev;
  __syncthreads();
  for (int off = 128; off > 0; off >>= 1) {
    if (tid < off) sRed[tid] += sRed[tid + off];
    __syncthreads();
  }
  const float inv = 1.0f / sRed[0];
  __syncthreads();
  if (tid < TP_) sScore[tid] = ev * inv;
  __syncthreads();

  // ---- out[b][e] = sum_t att[t] * keys[b][t][e]
  const int e    = tid & 63;
  const int part = tid >> 6;
  float acc = 0.0f;
  for (int t = part; t < T_; t += 4) acc += sScore[t] * kb[(size_t)t * 64 + e];
  sRed[tid] = acc;
  __syncthreads();
  if (tid < 64)
    out[(size_t)b * 64 + e] = sRed[e] + sRed[64 + e] + sRed[128 + e] + sRed[192 + e];
}

extern "C" void kernel_launch(void* const* d_in, const int* in_sizes, int n_in,
                              void* d_out, int out_size, void* d_ws, size_t ws_size,
                              hipStream_t stream) {
  const float* query = (const float*)d_in[0];
  const float* keys  = (const float*)d_in[1];
  const int*   klen  = (const int*)d_in[2];   // keys_length
  const float* W1    = (const float*)d_in[3];
  const float* b1    = (const float*)d_in[4];
  const float* a1    = (const float*)d_in[5];
  const float* W2    = (const float*)d_in[6];
  const float* b2    = (const float*)d_in[7];
  const float* a2    = (const float*)d_in[8];
  const float* W3    = (const float*)d_in[9];
  const float* b3    = (const float*)d_in[10];
  float* out = (float*)d_out;

  __bf16* wEffT = (__bf16*)d_ws;
  float*  wQ    = (float*)((char*)d_ws + 20480);
  __bf16* w2T   = (__bf16*)((char*)d_ws + 40960);

  const int B = in_sizes[0] / E_;   // 4096

  din_prep<<<1, 256, 0, stream>>>(W1, W2, wEffT, wQ, w2T);
  din_attn<<<B, 256, 0, stream>>>(query, keys, klen, b1, a1, b2, a2, W3, b3,
                                  wEffT, wQ, w2T, out);
}
